// ConvByMoveLayer_2662879724204
// MI455X (gfx1250) — matrix operand order repaired and checked by reading, not compiled, not run on hardware
//
#include <hip/hip_runtime.h>

// CDNA5 / gfx1250, wave32. D = A(16x32 f16) x B(32x16 f16) + C(16x16 f32)
typedef __attribute__((ext_vector_type(16))) _Float16 v16h;
typedef __attribute__((ext_vector_type(8)))  _Float16 v8h;
typedef __attribute__((ext_vector_type(8)))  float    v8f;
typedef __attribute__((ext_vector_type(4)))  float    v4f;

#define Bsz 128
#define Fsz 4096
#define Csz 32
#define Osz 32
#define Msz 9

__global__ __launch_bounds__(256) void conv_by_moves_wmma(
    const float* __restrict__ x,        // [B, F, C]
    const int*   __restrict__ mask,     // [M, F]
    const float* __restrict__ cc,       // [M, C, O]
    const float* __restrict__ biases,   // [F, O]
    float*       __restrict__ out)      // [B, F, O]
{
    // LDS: Wt[m][o][k] = cc[m][k][o], converted to f16 (9*32*32*2 = 18 KB)
    __shared__ _Float16 Wt[Msz * Csz * Osz];

    const int tid = threadIdx.x;
    for (int idx = tid; idx < Msz * Csz * Osz; idx += 256) {
        const int m = idx >> 10;         // /1024
        const int o = (idx >> 5) & 31;
        const int k = idx & 31;
        Wt[idx] = (_Float16)cc[(((m << 5) | k) << 5) | o];
    }
    __syncthreads();

    const int f       = blockIdx.x;
    const int wave    = tid >> 5;        // 8 waves: one 16-row batch tile each
    const int lane    = tid & 31;
    const int rowBase = wave << 4;       // batch tile origin
    const int row     = lane & 15;       // A-matrix row (M dim = batch)
    const int hi      = lane >> 4;       // half-wave select
    const int kbA     = hi << 4;         // A frag K base: 0 or 16
    const int kbB     = hi << 4;         // B frag K base: 0 or 16
    const int col     = lane & 15;       // B/C/D column (N dim = O)

    v8f acc0 = {};   // O columns  0..15
    v8f acc1 = {};   // O columns 16..31

#pragma unroll
    for (int m = 0; m < Msz; ++m) {
        const int src = mask[m * Fsz + f];   // uniform across the wave

        // A fragment: lane (row, half) reads one contiguous 16-float run of
        // x[rowBase+row, src, :]  (contiguous k = 16*half + s, matches B)
        const float* xr = x + (((size_t)(rowBase + row) * Fsz + (size_t)src) << 5);
        const v4f a0 = *(const v4f*)(xr + kbA);
        const v4f a1 = *(const v4f*)(xr + kbA + 4);
        const v4f a2 = *(const v4f*)(xr + kbA + 8);
        const v4f a3 = *(const v4f*)(xr + kbA + 12);

        v16h a;
#pragma unroll
        for (int j = 0; j < 4; ++j) {
            a[j]      = (_Float16)a0[j];
            a[4 + j]  = (_Float16)a1[j];
            a[8 + j]  = (_Float16)a2[j];
            a[12 + j] = (_Float16)a3[j];
        }

        // B fragments from LDS: 32 contiguous bytes per lane per O-half
        const _Float16* w0 = &Wt[(((m << 5) |  col)       << 5) | kbB];
        const _Float16* w1 = &Wt[(((m << 5) | (16 + col)) << 5) | kbB];
        v16h fb0, fb1;
#pragma unroll
        for (int j = 0; j < 16; ++j) {
            fb0[j] = w0[j];
            fb1[j] = w1[j];
        }

        // (neg_a, A, neg_b, B, c_mod, C, reuse_a, reuse_b)
        acc0 = __builtin_amdgcn_wmma_f32_16x16x32_f16(
                   false, a, false, fb0, (short)0, acc0, false, false);
        acc1 = __builtin_amdgcn_wmma_f32_16x16x32_f16(
                   false, a, false, fb1, (short)0, acc1, false, false);
    }

    // Bias + store per the f32 C/D layout: VGPR j -> row j (lanes 0-15),
    // row j+8 (lanes 16-31); N = lane&15.
    const float bia0 = biases[(f << 5) | col];
    const float bia1 = biases[(f << 5) | (16 + col)];
#pragma unroll
    for (int j = 0; j < 8; ++j) {
        const int bm = rowBase + j + (hi << 3);
        const size_t obase = ((size_t)bm * Fsz + (size_t)f) << 5;
        out[obase + col]      = acc0[j] + bia0;
        out[obase + 16 + col] = acc1[j] + bia1;
    }
}

extern "C" void kernel_launch(void* const* d_in, const int* in_sizes, int n_in,
                              void* d_out, int out_size, void* d_ws, size_t ws_size,
                              hipStream_t stream) {
    const float* x    = (const float*)d_in[0];
    const int*   mask = (const int*)  d_in[1];
    const float* cc   = (const float*)d_in[2];
    const float* bias = (const float*)d_in[3];
    float*       out  = (float*)d_out;

    conv_by_moves_wmma<<<dim3(Fsz), dim3(256), 0, stream>>>(x, mask, cc, bias, out);
}
